// MambaBlock_13812614824559
// MI455X (gfx1250) — matrix operand order repaired and checked by reading, not compiled, not run on hardware
//
#include <hip/hip_runtime.h>
#include <hip/hip_bf16.h>

// ---------- types ----------
typedef __attribute__((ext_vector_type(16))) __bf16  v16bf;
typedef __attribute__((ext_vector_type(8)))  float   v8f;
typedef __attribute__((ext_vector_type(4)))  unsigned int uvec4;

union FragAB { uvec4 u[2]; v16bf v; };

__device__ __forceinline__ unsigned short f2bf(float f) {
  unsigned int u = __float_as_uint(f);
  u += 0x7FFFu + ((u >> 16) & 1u);           // round-to-nearest-even
  return (unsigned short)(u >> 16);
}
__device__ __forceinline__ float bf2f(unsigned short h) {
  return __uint_as_float(((unsigned int)h) << 16);
}
__device__ __forceinline__ float sigm(float x) { return 1.0f / (1.0f + __expf(-x)); }
__device__ __forceinline__ float decay_a(float ld) {        // exp(-softplus(ld))
  float delta = (ld > 20.f) ? ld : log1pf(__expf(ld));
  return __expf(-delta);
}

// ---------- elementwise f32 -> bf16 ----------
__global__ void f32_to_bf16_kernel(const float* __restrict__ in,
                                   unsigned short* __restrict__ out, int n) {
  int i = blockIdx.x * blockDim.x + threadIdx.x;
  if (i < n) out[i] = f2bf(in[i]);
}

// ---------- layernorm (D=1024) -> bf16 ----------
__global__ __launch_bounds__(256) void layernorm_bf16_kernel(
    const float* __restrict__ x, const float* __restrict__ g,
    const float* __restrict__ b, unsigned short* __restrict__ out) {
  const int D = 1024;
  int row = blockIdx.x;
  const float* xr = x + (size_t)row * D;
  float v[4];
  float s = 0.f, s2 = 0.f;
#pragma unroll
  for (int i = 0; i < 4; ++i) {
    float t = xr[threadIdx.x + 256 * i];
    v[i] = t; s += t; s2 += t * t;
  }
  __shared__ float rs[256], rs2[256];
  rs[threadIdx.x] = s; rs2[threadIdx.x] = s2;
  __syncthreads();
  for (int off = 128; off > 0; off >>= 1) {
    if (threadIdx.x < off) {
      rs[threadIdx.x]  += rs[threadIdx.x + off];
      rs2[threadIdx.x] += rs2[threadIdx.x + off];
    }
    __syncthreads();
  }
  float mean = rs[0] * (1.f / D);
  float var  = rs2[0] * (1.f / D) - mean * mean;
  float rstd = rsqrtf(var + 1e-5f);
#pragma unroll
  for (int i = 0; i < 4; ++i) {
    int c = threadIdx.x + 256 * i;
    out[(size_t)row * D + c] = f2bf((v[i] - mean) * rstd * g[c] + b[c]);
  }
}

// ---------- WMMA GEMM:  C[M,N] = A[M,K](bf16) * B[N,K](bf16)^T + bias ----------
// Block tile 128x128, BK=64 (two 16x16x32 k-steps per LDS stage, 16 WMMA/stage).
// MODE 0: f32 out   MODE 1: bf16 out   MODE 2: f32 out + residual
template <int MODE>
__global__ __launch_bounds__(256) void wmma_gemm_kernel(
    const unsigned short* __restrict__ Abf, const unsigned short* __restrict__ Bbf,
    const float* __restrict__ bias, const float* __restrict__ resid,
    void* __restrict__ Cout, int M, int N, int K) {
  __shared__ __align__(16) unsigned short sA[128 * 72];   // 18 KB (stride 144B)
  __shared__ __align__(16) unsigned short sB[128 * 72];   // 18 KB

  const int tid   = threadIdx.x;
  const int lane  = tid & 31;
  const int wave  = tid >> 5;        // 0..7
  const int waveM = wave & 1;        // 2 waves along M
  const int waveN = wave >> 1;       // 4 waves along N
  const int m0 = waveM * 64;
  const int n0 = waveN * 32;

  // global->LDS staging: each thread moves 32 bf16 of A and of B per stage
  const int ldRow = tid >> 1;            // 0..127
  const int ldCol = (tid & 1) * 32;      // 0 or 32
  const size_t aBase = ((size_t)(blockIdx.y * 128 + ldRow)) * K + ldCol;
  const size_t bBase = ((size_t)(blockIdx.x * 128 + ldRow)) * K + ldCol;

  v8f acc[4][2] = {};

  for (int k0 = 0; k0 < K; k0 += 64) {
    uvec4 ra[4], rb[4];
#pragma unroll
    for (int q = 0; q < 4; ++q) {
      ra[q] = *(const uvec4*)(Abf + aBase + k0 + 8 * q);
      rb[q] = *(const uvec4*)(Bbf + bBase + k0 + 8 * q);
    }
    __syncthreads();                      // previous stage's frag reads done
#pragma unroll
    for (int q = 0; q < 4; ++q) {
      *(uvec4*)(sA + ldRow * 72 + ldCol + 8 * q) = ra[q];
      *(uvec4*)(sB + ldRow * 72 + ldCol + 8 * q) = rb[q];
    }
    __syncthreads();

#pragma unroll
    for (int kk = 0; kk < 64; kk += 32) {
      // A fragments (16x32): lane<16 -> M=lane, K chunks {0..7,16..23}; lane>=16 -> {8..15,24..31}
      FragAB fa[4];
#pragma unroll
      for (int i = 0; i < 4; ++i) {
        const unsigned short* p =
            sA + (m0 + i * 16 + (lane & 15)) * 72 + kk + ((lane >> 4) * 8);
        fa[i].u[0] = *(const uvec4*)p;
        fa[i].u[1] = *(const uvec4*)(p + 16);
      }
      // B fragments (32x16): lane%16 = column, lane<16 -> K chunks {0..7,16..23}; lane>=16 -> {8..15,24..31}
      FragAB fb[2];
#pragma unroll
      for (int j = 0; j < 2; ++j) {
        const unsigned short* p =
            sB + (n0 + j * 16 + (lane & 15)) * 72 + kk + ((lane >> 4) * 8);
        fb[j].u[0] = *(const uvec4*)p;
        fb[j].u[1] = *(const uvec4*)(p + 16);
      }
#pragma unroll
      for (int i = 0; i < 4; ++i)
#pragma unroll
        for (int j = 0; j < 2; ++j)
          acc[i][j] = __builtin_amdgcn_wmma_f32_16x16x32_bf16(
              false, fa[i].v, false, fb[j].v, (short)0, acc[i][j], false, false);
    }
  }

  // C layout: VGPR r -> M=r (lanes 0-15) / M=r+8 (lanes 16-31), N = lane%16
  const int rowBase = blockIdx.y * 128 + m0 + ((lane >> 4) * 8);
  const int colBase = blockIdx.x * 128 + n0 + (lane & 15);
#pragma unroll
  for (int i = 0; i < 4; ++i) {
#pragma unroll
    for (int j = 0; j < 2; ++j) {
      int col = colBase + j * 16;
      float bv = bias[col];
#pragma unroll
      for (int r = 0; r < 8; ++r) {
        int row = rowBase + i * 16 + r;
        size_t idx = (size_t)row * N + col;
        float v = acc[i][j][r] + bv;
        if (MODE == 0)      ((float*)Cout)[idx] = v;
        else if (MODE == 1) ((unsigned short*)Cout)[idx] = f2bf(v);
        else                ((float*)Cout)[idx] = v + resid[idx];
      }
    }
  }
}

// ---------- causal depthwise conv (K=4) + SiLU ----------
__global__ void conv_silu_kernel(const unsigned short* __restrict__ xproj,
                                 const float* __restrict__ w, const float* __restrict__ cb,
                                 unsigned short* __restrict__ xc) {
  int idx = blockIdx.x * blockDim.x + threadIdx.x;  // 8192*2048 exact
  int e = idx & 2047;
  int t = idx >> 11;      // token id
  int s = t & 2047;       // position within sequence
  float acc = cb[e];
#pragma unroll
  for (int k = 0; k < 4; ++k) {
    int sp = s + k - 3;
    if (sp >= 0)
      acc += w[e * 4 + k] * bf2f(xproj[(size_t)(t + k - 3) * 4096 + e]);
  }
  xc[idx] = f2bf(acc * sigm(acc));
}

// ---------- u = c * sigmoid(d), in place into c ----------
__global__ void glu_combine_kernel(float* __restrict__ c, const float* __restrict__ d, int n) {
  int i = blockIdx.x * blockDim.x + threadIdx.x;
  if (i < n) c[i] = c[i] * sigm(d[i]);
}

// ---------- chunked SSM scan (S=2048 split into 8 chunks of 256) ----------
// Pass A: per (chunk,b,e) local carry with zero initial state
__global__ void scan_carry_kernel(const float* __restrict__ u,
                                  const float* __restrict__ log_delta,
                                  float* __restrict__ carry) {
  int tid = blockIdx.x * blockDim.x + threadIdx.x;  // 8*4*2048 = 65536
  int e     = tid & 2047;
  int b     = (tid >> 11) & 3;
  int chunk = tid >> 13;
  float a = decay_a(log_delta[e]);
  float s = 0.f;
  const size_t tok0 = (size_t)b * 2048 + chunk * 256;
  for (int t = 0; t < 256; ++t)
    s = a * s + u[(tok0 + t) * 2048 + e];
  carry[(size_t)chunk * 8192 + b * 2048 + e] = s;
}

// Pass B: serial combine across 8 chunks per (b,e); prefix = state entering chunk
__global__ void scan_prefix_kernel(const float* __restrict__ carry,
                                   const float* __restrict__ log_delta,
                                   float* __restrict__ prefix) {
  int tid = blockIdx.x * blockDim.x + threadIdx.x;  // 8192
  int e  = tid & 2047;
  float ld = log_delta[e];
  float delta = (ld > 20.f) ? ld : log1pf(__expf(ld));
  float a256 = __expf(-256.f * delta);              // a^256
  float p = 0.f;
#pragma unroll
  for (int c = 0; c < 8; ++c) {
    prefix[(size_t)c * 8192 + tid] = p;
    p = a256 * p + carry[(size_t)c * 8192 + tid];
  }
}

// Pass C: re-scan each chunk from prefix, fuse gate -> xg bf16
__global__ void scan_apply_kernel(const float* __restrict__ u,
                                  const unsigned short* __restrict__ xproj,
                                  const float* __restrict__ Amat, const float* __restrict__ Bm,
                                  const float* __restrict__ log_delta,
                                  const float* __restrict__ prefix,
                                  unsigned short* __restrict__ xg) {
  int tid = blockIdx.x * blockDim.x + threadIdx.x;  // 65536
  int e     = tid & 2047;
  int b     = (tid >> 11) & 3;
  int chunk = tid >> 13;
  float a = decay_a(log_delta[e]);
  float c = 0.f;
#pragma unroll
  for (int n = 0; n < 16; ++n) c += Amat[e * 16 + n] * Bm[e * 16 + n];
  float s = prefix[(size_t)chunk * 8192 + b * 2048 + e];
  const size_t tok0 = (size_t)b * 2048 + chunk * 256;
  for (int t = 0; t < 256; ++t) {
    size_t tok = tok0 + t;
    s = a * s + u[tok * 2048 + e];
    float g = bf2f(xproj[tok * 4096 + 2048 + e]);   // gate half of x_proj
    xg[tok * 2048 + e] = f2bf(c * s * sigm(g));
  }
}

// ---------- host orchestration ----------
extern "C" void kernel_launch(void* const* d_in, const int* in_sizes, int n_in,
                              void* d_out, int out_size, void* d_ws, size_t ws_size,
                              hipStream_t stream) {
  const float* x       = (const float*)d_in[0];
  const float* ln_g    = (const float*)d_in[1];
  const float* ln_b    = (const float*)d_in[2];
  const float* W_in    = (const float*)d_in[3];
  const float* b_in    = (const float*)d_in[4];
  const float* conv_w  = (const float*)d_in[5];
  const float* conv_b  = (const float*)d_in[6];
  const float* Amat    = (const float*)d_in[7];
  const float* Bm      = (const float*)d_in[8];
  const float* W_c     = (const float*)d_in[9];
  const float* b_c     = (const float*)d_in[10];
  const float* W_d     = (const float*)d_in[11];
  const float* b_d     = (const float*)d_in[12];
  const float* log_dl  = (const float*)d_in[13];
  const float* W_out   = (const float*)d_in[14];
  const float* b_out   = (const float*)d_in[15];
  float* out = (float*)d_out;

  char* ws = (char*)d_ws;
  unsigned short* xln   = (unsigned short*)(ws + 0);          // 8192*1024 bf16
  unsigned short* winb  = (unsigned short*)(ws + 16777216);   // 4096*1024 bf16
  unsigned short* wcb   = (unsigned short*)(ws + 25165824);   // 2048*2048 bf16
  unsigned short* wdb   = (unsigned short*)(ws + 33554432);   // 2048*2048 bf16
  unsigned short* woutb = (unsigned short*)(ws + 41943040);   // 1024*2048 bf16
  unsigned short* xproj = (unsigned short*)(ws + 46137344);   // 8192*4096 bf16
  unsigned short* xc    = (unsigned short*)(ws + 113246208);  // 8192*2048 bf16 (reused as xg)
  float*          cbuf  = (float*)(ws + 146800640);           // 8192*2048 f32 (becomes u)
  float*          dbuf  = (float*)(ws + 213909504);           // 8192*2048 f32
  float*          carry = (float*)(ws + 281018368);           // 8*8192 f32
  float*          prefx = (float*)(ws + 281280512);           // 8*8192 f32

  // weights -> bf16
  f32_to_bf16_kernel<<<16384, 256, 0, stream>>>(W_in,  winb,  4194304);
  f32_to_bf16_kernel<<<16384, 256, 0, stream>>>(W_c,   wcb,   4194304);
  f32_to_bf16_kernel<<<16384, 256, 0, stream>>>(W_d,   wdb,   4194304);
  f32_to_bf16_kernel<<<8192,  256, 0, stream>>>(W_out, woutb, 2097152);

  // layernorm -> bf16
  layernorm_bf16_kernel<<<8192, 256, 0, stream>>>(x, ln_g, ln_b, xln);

  // GEMM1: x_proj = xln @ W_in^T + b_in  -> bf16 (M=8192, N=4096, K=1024)
  wmma_gemm_kernel<1><<<dim3(32, 64), 256, 0, stream>>>(xln, winb, b_in, nullptr,
                                                        xproj, 8192, 4096, 1024);
  // causal conv + SiLU on first half of x_proj -> xc bf16
  conv_silu_kernel<<<65536, 256, 0, stream>>>(xproj, conv_w, conv_b, xc);

  // GEMM2/3: c_pre, d_pre  (M=8192, N=2048, K=2048)
  wmma_gemm_kernel<0><<<dim3(16, 64), 256, 0, stream>>>(xc, wcb, b_c, nullptr,
                                                        cbuf, 8192, 2048, 2048);
  wmma_gemm_kernel<0><<<dim3(16, 64), 256, 0, stream>>>(xc, wdb, b_d, nullptr,
                                                        dbuf, 8192, 2048, 2048);
  // u = c_pre * sigmoid(d_pre)  (in place in cbuf)
  glu_combine_kernel<<<65536, 256, 0, stream>>>(cbuf, dbuf, 16777216);

  // chunked SSM scan + gating -> xg bf16 (reuse xc buffer)
  scan_carry_kernel <<<256, 256, 0, stream>>>(cbuf, log_dl, carry);
  scan_prefix_kernel<<<32,  256, 0, stream>>>(carry, log_dl, prefx);
  scan_apply_kernel <<<256, 256, 0, stream>>>(cbuf, xproj, Amat, Bm, log_dl, prefx, xc);

  // GEMM4: out = xg @ W_out^T + b_out + x  (M=8192, N=1024, K=2048)
  wmma_gemm_kernel<2><<<dim3(8, 64), 256, 0, stream>>>(xc, woutb, b_out, x,
                                                       out, 8192, 1024, 2048);
}
